// CausalSelfAttention_76742475645091
// MI455X (gfx1250) — matrix operand order repaired and checked by reading, not compiled, not run on hardware
//
#include <hip/hip_runtime.h>
#include <hip/hip_bf16.h>

// Problem constants (match reference)
#define B_  4
#define T_  2048
#define C_  1024
#define H_  16
#define HD_ 64
#define BT_ (B_ * T_)   // 8192

typedef __attribute__((ext_vector_type(16))) __bf16 v16bf;
typedef __attribute__((ext_vector_type(8)))  __bf16 v8bf;
typedef __attribute__((ext_vector_type(8)))  float  v8f;

__device__ __forceinline__ __bf16 f2bf(float f) {
    unsigned u = __builtin_bit_cast(unsigned, f);
    unsigned r = u + 0x7fffu + ((u >> 16) & 1u);   // round-to-nearest-even
    unsigned short h = (unsigned short)(r >> 16);
    return __builtin_bit_cast(__bf16, h);
}

__device__ __forceinline__ v8f wmma_bf16(v16bf a, v16bf b, v8f c) {
    // D = A(16x32 bf16) * B(32x16 bf16) + C(16x16 f32)
    return __builtin_amdgcn_wmma_f32_16x16x32_bf16(
        /*neg_a=*/false, a, /*neg_b=*/false, b,
        /*c_mod=*/(short)0, c, /*reuse_a=*/false, /*reuse_b=*/false);
}

__device__ __forceinline__ void wait_asynccnt0() {
    asm volatile("s_wait_asynccnt 0x0" ::: "memory");
}

// A-operand (16x32, row-major M x K in memory):
// lane: M = lane&15, half = lane>>4
// VGPR 0..3 : K = half*16 + {0..7}     (contiguous 8 bf16)
// VGPR 4..7 : K = half*16 + 8 + {0..7} (contiguous 8 bf16)
__device__ __forceinline__ v16bf load_a_frag(const __bf16* base, int ld, int row, int kbase, int lane) {
    int m = lane & 15, half = lane >> 4;
    const __bf16* p = base + (size_t)(row + m) * ld + kbase + half * 16;
    v8bf lo = *(const v8bf*)p;
    v8bf hi = *(const v8bf*)(p + 8);
    v16bf a;
#pragma unroll
    for (int i = 0; i < 8; ++i) { a[i] = lo[i]; a[i + 8] = hi[i]; }
    return a;
}

// B-operand (32x16): memory holds B^T row-major (N rows x K cols).
// lane: N = lane&15, half = lane>>4 ; K = half*16 + {0..15} (contiguous 16 bf16)
__device__ __forceinline__ v16bf load_b_frag(const __bf16* base, int ld, int nrow, int kbase, int lane) {
    int n = lane & 15, half = lane >> 4;
    const __bf16* p = base + (size_t)(nrow + n) * ld + kbase + half * 16;
    return *(const v16bf*)p;
}

// Same B-fragment pattern, from a staged LDS tile [64 rows][64 K]
__device__ __forceinline__ v16bf load_b_frag_lds64(const __bf16* tile, int nrow, int ksub, int lane) {
    int n = lane & 15, half = lane >> 4;
    const __bf16* p = tile + (nrow + n) * 64 + ksub + half * 16;
    return *(const v16bf*)p;     // -> ds_load_b128 x2
}

// Cooperatively stage a 64(N) x 64(K) bf16 tile of W (N x K row-major) into LDS
// via CDNA5 async copies (ASYNCcnt). 128 threads: thread t copies 64 bytes
// (half a 128-byte row) using one base address + INST_OFFSET 0/16/32/48
// (INST_OFFSET is added to BOTH the LDS and global addresses per the ISA).
__device__ __forceinline__ void stage_b_tile64(__bf16* dst, const __bf16* w, int n0, int kb, int tid) {
    int row = tid >> 1;                 // 0..63
    int col = (tid & 1) * 32;           // element offset within 64-elem (128B) row
    const __bf16* src = w + (size_t)(n0 + row) * C_ + kb + col;
    __bf16* d = dst + row * 64 + col;
    unsigned loff = (unsigned)(size_t)d;                 // low 32 bits = LDS byte offset
    unsigned long long ga = (unsigned long long)(size_t)src;
    asm volatile("global_load_async_to_lds_b128 %0, %1, off"           :: "v"(loff), "v"(ga) : "memory");
    asm volatile("global_load_async_to_lds_b128 %0, %1, off offset:16" :: "v"(loff), "v"(ga) : "memory");
    asm volatile("global_load_async_to_lds_b128 %0, %1, off offset:32" :: "v"(loff), "v"(ga) : "memory");
    asm volatile("global_load_async_to_lds_b128 %0, %1, off offset:48" :: "v"(loff), "v"(ga) : "memory");
}

// ---------------------------------------------------------------- conversion
__global__ void f32_to_bf16_kernel(const float* __restrict__ src, __bf16* __restrict__ dst, int n) {
    int i = blockIdx.x * blockDim.x + threadIdx.x;
    int stride = gridDim.x * blockDim.x;
    for (; i < n; i += stride) dst[i] = f2bf(src[i]);
}

// ---------------------------------------------------------------- QKV projection
// y = x @ W^T + b ; W is (Cout, Cin) row-major, which IS the N x K layout we need.
// grid = (C/64, BT/64, 3), block = 128 (4 waves); wave computes 16 rows x 64 cols.
// B tile (shared by all 4 waves) is double-buffered in LDS via async copies;
// one block barrier per 8 WMMAs.
__global__ __launch_bounds__(128)
void qkv_proj_kernel(const __bf16* __restrict__ xb,
                     const __bf16* __restrict__ wq, const __bf16* __restrict__ wk,
                     const __bf16* __restrict__ wv,
                     const float* __restrict__ bq, const float* __restrict__ bk,
                     const float* __restrict__ bv,
                     __bf16* __restrict__ q, __bf16* __restrict__ k, __bf16* __restrict__ vt) {
    __shared__ __align__(32) __bf16 bstage[2][64 * 64];   // 2 x 8KB

    int tid = threadIdx.x;
    int lane = tid & 31;
    int wave = tid >> 5;
    int m0 = blockIdx.y * 64 + wave * 16;
    int n0 = blockIdx.x * 64;
    int which = blockIdx.z;
    const __bf16* w = (which == 0) ? wq : (which == 1) ? wk : wv;
    const float* bias = (which == 0) ? bq : (which == 1) ? bk : bv;

    int cur = 0;
    stage_b_tile64(bstage[0], w, n0, 0, tid);
    wait_asynccnt0();
    __syncthreads();

    v8f c[4] = {};
    for (int kb = 0; kb < C_; kb += 64) {
        if (kb + 64 < C_) stage_b_tile64(bstage[cur ^ 1], w, n0, kb + 64, tid);
#pragma unroll
        for (int ks = 0; ks < 2; ++ks) {
            v16bf a = load_a_frag(xb, C_, m0, kb + 32 * ks, lane);
#pragma unroll
            for (int t = 0; t < 4; ++t) {
                v16bf bf = load_b_frag_lds64(bstage[cur], 16 * t, 32 * ks, lane);
                c[t] = wmma_bf16(a, bf, c[t]);
            }
        }
        wait_asynccnt0();
        __syncthreads();
        cur ^= 1;
    }

    int half = lane >> 4, nl = lane & 15;
#pragma unroll
    for (int t = 0; t < 4; ++t) {
#pragma unroll
        for (int r = 0; r < 8; ++r) {
            int m = m0 + r + half * 8;          // global row in [0, B*T)
            int n = n0 + 16 * t + nl;           // global col in [0, C)
            float val = c[t][r] + bias[n];
            int b = m >> 11;                    // /T (T=2048)
            int tt = m & (T_ - 1);
            int h = n >> 6;                     // /HD
            int d = n & (HD_ - 1);
            __bf16 bv16 = f2bf(val);
            if (which < 2) {
                __bf16* dst = (which == 0) ? q : k;
                dst[(((size_t)b * H_ + h) * T_ + tt) * HD_ + d] = bv16;   // [B,H,T,HD]
            } else {
                vt[(((size_t)b * H_ + h) * HD_ + d) * T_ + tt] = bv16;    // [B,H,HD,T]
            }
        }
    }
}

// ---------------------------------------------------------------- flash attention
// grid = (T/64, B*H), block = 128 (4 waves); wave owns 16 query rows.
__global__ __launch_bounds__(128)
void flash_attn_kernel(const __bf16* __restrict__ q, const __bf16* __restrict__ kk,
                       const __bf16* __restrict__ vt, const int* __restrict__ amask,
                       __bf16* __restrict__ y) {
    __shared__ __align__(16) __bf16 pls[4][16][32];   // wave-private P tiles

    int lane = threadIdx.x & 31;
    int wave = threadIdx.x >> 5;
    int bh = blockIdx.y;
    int b = bh >> 4, h = bh & (H_ - 1);
    int q0 = blockIdx.x * 64 + wave * 16;

    const __bf16* qh = q  + (size_t)bh * T_ * HD_;
    const __bf16* kh = kk + (size_t)bh * T_ * HD_;
    const __bf16* vh = vt + (size_t)bh * HD_ * T_;    // rows = d, ld = T
    const int*  mrow = amask + (size_t)b * T_;

    // Q fragments for the two K-steps of HD=64
    v16bf aq0 = load_a_frag(qh, HD_, q0, 0, lane);
    v16bf aq1 = load_a_frag(qh, HD_, q0, 32, lane);

    float mrun[8], lrun[8];
#pragma unroll
    for (int r = 0; r < 8; ++r) { mrun[r] = -1e30f; lrun[r] = 0.0f; }
    v8f o[4] = {};

    int half = lane >> 4, nl = lane & 15;
    const float scale = 0.125f;   // 1/sqrt(64)

    for (int j0 = 0; j0 < q0 + 16; j0 += 32) {
        // S = Q * K^T over this 32-key tile (two 16-col fragments)
        v8f s[2] = {};
#pragma unroll
        for (int nt = 0; nt < 2; ++nt) {
            v16bf bk0 = load_b_frag(kh, HD_, j0 + 16 * nt, 0, lane);
            s[nt] = wmma_bf16(aq0, bk0, s[nt]);
            v16bf bk1 = load_b_frag(kh, HD_, j0 + 16 * nt, 32, lane);
            s[nt] = wmma_bf16(aq1, bk1, s[nt]);
        }
        // scale + causal + padding mask
#pragma unroll
        for (int nt = 0; nt < 2; ++nt) {
            int key = j0 + 16 * nt + nl;
            bool ok = (mrow[key] != 0);
#pragma unroll
            for (int r = 0; r < 8; ++r) {
                int qi = q0 + r + half * 8;
                float sv = s[nt][r] * scale;
                if (key > qi || !ok) sv = -1e30f;
                s[nt][r] = sv;
            }
        }
        // online softmax; row M lives across the 16 lanes of one half-wave
#pragma unroll
        for (int r = 0; r < 8; ++r) {
            float mx = fmaxf(s[0][r], s[1][r]);
            mx = fmaxf(mx, __shfl_xor(mx, 1, 32));
            mx = fmaxf(mx, __shfl_xor(mx, 2, 32));
            mx = fmaxf(mx, __shfl_xor(mx, 4, 32));
            mx = fmaxf(mx, __shfl_xor(mx, 8, 32));
            float mnew = fmaxf(mrun[r], mx);
            float corr = __expf(mrun[r] - mnew);
            mrun[r] = mnew;
            float p0 = __expf(s[0][r] - mnew);
            float p1 = __expf(s[1][r] - mnew);
            float rs = p0 + p1;
            rs += __shfl_xor(rs, 1, 32);
            rs += __shfl_xor(rs, 2, 32);
            rs += __shfl_xor(rs, 4, 32);
            rs += __shfl_xor(rs, 8, 32);
            lrun[r] = lrun[r] * corr + rs;
#pragma unroll
            for (int dt = 0; dt < 4; ++dt) o[dt][r] = o[dt][r] * corr;
            // D-layout -> LDS (row-major 16x32 P tile)
            int m = r + half * 8;
            pls[wave][m][nl]      = f2bf(p0);
            pls[wave][m][16 + nl] = f2bf(p1);
        }
        // reload P as an A-fragment (K = 32 keys)
        v16bf ap;
        {
            const __bf16* pp = &pls[wave][nl][0];
            v8bf lo = *(const v8bf*)(pp + half * 16);
            v8bf hi = *(const v8bf*)(pp + half * 16 + 8);
#pragma unroll
            for (int i = 0; i < 8; ++i) { ap[i] = lo[i]; ap[i + 8] = hi[i]; }
        }
        // O += P * V  (B-operand = V^T rows, contiguous over keys)
#pragma unroll
        for (int dt = 0; dt < 4; ++dt) {
            v16bf bv = load_b_frag(vh, T_, 16 * dt, j0, lane);
            o[dt] = wmma_bf16(ap, bv, o[dt]);
        }
    }

    // normalize and write y[B,T,C] (bf16, feeds the output projection)
#pragma unroll
    for (int r = 0; r < 8; ++r) {
        float inv = 1.0f / lrun[r];
        int qi = q0 + r + half * 8;
#pragma unroll
        for (int dt = 0; dt < 4; ++dt) {
            int d = 16 * dt + nl;
            y[((size_t)b * T_ + qi) * C_ + h * HD_ + d] = f2bf(o[dt][r] * inv);
        }
    }
}

// ---------------------------------------------------------------- output projection
// out = y @ Wp^T + bp (fp32 out). grid = (C/64, BT/64), block = 128.
__global__ __launch_bounds__(128)
void out_proj_kernel(const __bf16* __restrict__ yb, const __bf16* __restrict__ wp,
                     const float* __restrict__ bp, float* __restrict__ out) {
    __shared__ __align__(32) __bf16 bstage[2][64 * 64];   // 2 x 8KB

    int tid = threadIdx.x;
    int lane = tid & 31;
    int wave = tid >> 5;
    int m0 = blockIdx.y * 64 + wave * 16;
    int n0 = blockIdx.x * 64;

    int cur = 0;
    stage_b_tile64(bstage[0], wp, n0, 0, tid);
    wait_asynccnt0();
    __syncthreads();

    v8f c[4] = {};
    for (int kb = 0; kb < C_; kb += 64) {
        if (kb + 64 < C_) stage_b_tile64(bstage[cur ^ 1], wp, n0, kb + 64, tid);
#pragma unroll
        for (int ks = 0; ks < 2; ++ks) {
            v16bf a = load_a_frag(yb, C_, m0, kb + 32 * ks, lane);
#pragma unroll
            for (int t = 0; t < 4; ++t) {
                v16bf bf = load_b_frag_lds64(bstage[cur], 16 * t, 32 * ks, lane);
                c[t] = wmma_bf16(a, bf, c[t]);
            }
        }
        wait_asynccnt0();
        __syncthreads();
        cur ^= 1;
    }

    int half = lane >> 4, nl = lane & 15;
#pragma unroll
    for (int t = 0; t < 4; ++t) {
#pragma unroll
        for (int r = 0; r < 8; ++r) {
            int m = m0 + r + half * 8;
            int n = n0 + 16 * t + nl;
            out[(size_t)m * C_ + n] = c[t][r] + bp[n];
        }
    }
}

// ---------------------------------------------------------------- launch
extern "C" void kernel_launch(void* const* d_in, const int* in_sizes, int n_in,
                              void* d_out, int out_size, void* d_ws, size_t ws_size,
                              hipStream_t stream) {
    const float* x  = (const float*)d_in[0];
    const float* Wq = (const float*)d_in[1];
    const float* bq = (const float*)d_in[2];
    const float* Wk = (const float*)d_in[3];
    const float* bk = (const float*)d_in[4];
    const float* Wv = (const float*)d_in[5];
    const float* bv = (const float*)d_in[6];
    const float* Wp = (const float*)d_in[7];
    const float* bp = (const float*)d_in[8];
    const int* amask = (const int*)d_in[9];

    // workspace layout (bf16 intermediates), total ~88 MB
    char* ws = (char*)d_ws;
    __bf16* xb  = (__bf16*)ws; ws += (size_t)BT_ * C_ * 2;
    __bf16* wqb = (__bf16*)ws; ws += (size_t)C_ * C_ * 2;
    __bf16* wkb = (__bf16*)ws; ws += (size_t)C_ * C_ * 2;
    __bf16* wvb = (__bf16*)ws; ws += (size_t)C_ * C_ * 2;
    __bf16* wpb = (__bf16*)ws; ws += (size_t)C_ * C_ * 2;
    __bf16* qb  = (__bf16*)ws; ws += (size_t)BT_ * C_ * 2;
    __bf16* kb  = (__bf16*)ws; ws += (size_t)BT_ * C_ * 2;
    __bf16* vtb = (__bf16*)ws; ws += (size_t)BT_ * C_ * 2;
    __bf16* yb  = (__bf16*)ws; ws += (size_t)BT_ * C_ * 2;

    // stage 0: fp32 -> bf16
    f32_to_bf16_kernel<<<2048, 256, 0, stream>>>(x, xb, BT_ * C_);
    f32_to_bf16_kernel<<<1024, 256, 0, stream>>>(Wq, wqb, C_ * C_);
    f32_to_bf16_kernel<<<1024, 256, 0, stream>>>(Wk, wkb, C_ * C_);
    f32_to_bf16_kernel<<<1024, 256, 0, stream>>>(Wv, wvb, C_ * C_);
    f32_to_bf16_kernel<<<1024, 256, 0, stream>>>(Wp, wpb, C_ * C_);

    // stage 1: QKV projections (z selects q/k/v)
    qkv_proj_kernel<<<dim3(C_ / 64, BT_ / 64, 3), 128, 0, stream>>>(
        xb, wqb, wkb, wvb, bq, bk, bv, qb, kb, vtb);

    // stage 2: causal flash attention
    flash_attn_kernel<<<dim3(T_ / 64, B_ * H_), 128, 0, stream>>>(qb, kb, vtb, amask, yb);

    // stage 3: output projection -> fp32 d_out
    out_proj_kernel<<<dim3(C_ / 64, BT_ / 64), 128, 0, stream>>>(yb, wpb, bp, (float*)d_out);
}
